// DirectionalPropagation_52390011077094
// MI455X (gfx1250) — matrix operand order repaired and checked by reading, not compiled, not run on hardware
//
#include <hip/hip_runtime.h>
#include <hip/hip_bf16.h>
#include <math.h>

typedef __attribute__((ext_vector_type(16))) _Float16 v16h;
typedef __attribute__((ext_vector_type(8)))  _Float16 v8h;
typedef __attribute__((ext_vector_type(8)))  float    v8f;

#define HDIM 128
#define TDIM 32
#define MDIM 16

// ---------------- f32 -> f16 conversion of node features (L2-resident) ------
__global__ void cvt_f16_kernel(const float* __restrict__ x,
                               _Float16* __restrict__ xh, int n) {
  for (int i = blockIdx.x * blockDim.x + threadIdx.x; i < n;
       i += gridDim.x * blockDim.x)
    xh[i] = (_Float16)x[i];
}

// ---------------- W_trans [256,32] -> Wt [32][256] f16 (B-fragment friendly) -
__global__ void transpose_w_kernel(const float* __restrict__ W,
                                   _Float16* __restrict__ Wt) {
  int i = blockIdx.x * blockDim.x + threadIdx.x;
  if (i < 2 * HDIM * TDIM) {
    int c = i / (2 * HDIM);
    int k = i % (2 * HDIM);
    Wt[c * (2 * HDIM) + k] = (_Float16)W[k * TDIM + c];
  }
}

// ---------------- per-edge GEMM + tanh + gate dot + sigmoid ------------------
// One wave = one 16-edge tile. D = A(16x256 gathered f16) * B(256x32 Wt) via
// 8 chained v_wmma_f32_16x16x32_f16 per 16-col block (2 blocks).
// B (Wt, 16KB) is staged in LDS once per block: B-fragment reads ride the DS
// pipe and co-issue with the random global A-gathers.
__global__ __launch_bounds__(256) void edge_gemm_kernel(
    const _Float16* __restrict__ xh, const _Float16* __restrict__ Wt,
    const int* __restrict__ eidx, int E, const float* __restrict__ eattr,
    int ad, const float* __restrict__ Wp, const float* __restrict__ bp,
    const float* __restrict__ btrans, float* __restrict__ ew) {
  __shared__ _Float16 sW[TDIM * 2 * HDIM];  // 16 KB of 320 KB WGP LDS

  // cooperative stage: 1024 x 16B chunks, 256 threads (global b128 -> ds b128)
  {
    const float4* src = (const float4*)Wt;
    float4* dst = (float4*)sW;
    const int chunks = (TDIM * 2 * HDIM * (int)sizeof(_Float16)) / 16;
#pragma unroll
    for (int i = threadIdx.x; i < chunks; i += 256) dst[i] = src[i];
  }
  __syncthreads();

  const int lane = threadIdx.x & 31;
  const int wave = threadIdx.x >> 5;
  const long long t0 = ((long long)blockIdx.x * 8 + wave) * 16;
  if (t0 >= E) return;  // wave-uniform: EXEC stays all-ones for WMMA

  const int lsub = lane & 15;
  const int half = lane >> 4;

  // prefetch next tile's indices into GL2 (emits global_prefetch_b8)
  long long pf = t0 + 128; if (pf >= E) pf = t0;
  __builtin_prefetch(eidx + pf, 0, 3);

  long long eM = t0 + lsub;            // A-matrix row M = lane%16
  if (eM >= E) eM = E - 1;             // clamp (loads only; stores guarded)
  const int sidx = eidx[eM];
  const int didx = eidx[(long long)E + eM];
  const _Float16* rs = xh + (long long)sidx * HDIM;
  const _Float16* rd = xh + (long long)didx * HDIM;

  const int col0 = lsub;               // output columns this lane holds
  const int col1 = 16 + lsub;
  const _Float16* wb0 = sW + col0 * (2 * HDIM) + half * 16;
  const _Float16* wb1 = sW + col1 * (2 * HDIM) + half * 16;

  union U { v16h v; v8h h[2]; };
  v8f acc0 = {};
  v8f acc1 = {};

#pragma unroll
  for (int s = 0; s < 8; ++s) {
    const int base = 32 * s;
    // A fragment: low half-wave K {0..15}, high half {16..31}
    const int off0 = base + (half ? 16 : 0);
    const int off1 = off0 + 8;
    U a;
    a.h[0] = *(const v8h*)((off0 < HDIM) ? (rs + off0) : (rd + (off0 - HDIM)));
    a.h[1] = *(const v8h*)((off1 < HDIM) ? (rs + off1) : (rd + (off1 - HDIM)));
    // B fragments from LDS: lane n=lane%16, K run of 16 at 32*s + 16*half
    U b0, b1;
    b0.h[0] = *(const v8h*)(wb0 + base);
    b0.h[1] = *(const v8h*)(wb0 + base + 8);
    b1.h[0] = *(const v8h*)(wb1 + base);
    b1.h[1] = *(const v8h*)(wb1 + base + 8);
    acc0 = __builtin_amdgcn_wmma_f32_16x16x32_f16(false, a.v, false, b0.v,
                                                  (short)0, acc0, false, false);
    acc1 = __builtin_amdgcn_wmma_f32_16x16x32_f16(false, a.v, false, b1.v,
                                                  (short)0, acc1, false, false);
  }

  // tanh + gate weight contribution; reduce 16 columns across each half-wave.
  const float wA = Wp[ad + col0];
  const float wB = Wp[ad + col1];
  const float bt0 = btrans[col0];
  const float bt1 = btrans[col1];

  float sums[8];
#pragma unroll
  for (int r = 0; r < 8; ++r) {
    float c = tanhf(acc0[r] + bt0) * wA + tanhf(acc1[r] + bt1) * wB;
    c += __shfl_xor(c, 1, 16);
    c += __shfl_xor(c, 2, 16);
    c += __shfl_xor(c, 4, 16);
    c += __shfl_xor(c, 8, 16);
    sums[r] = c;   // row sums for rows (half*8 + r), replicated in half-wave
  }

  if (lsub < 8) {
    const int M = half * 8 + lsub;
    const long long e = t0 + M;
    if (e < E) {
      float t = sums[lsub] + bp[0];
      for (int j = 0; j < ad; ++j) t += eattr[e * ad + j] * Wp[j];
      ew[e] = 1.0f / (1.0f + __expf(-t));
    }
  }
}

// ---------------- mask copy ---------------------------------------------------
__global__ void copy_f32_kernel(const float* __restrict__ in,
                                float* __restrict__ out, int n) {
  for (int i = blockIdx.x * blockDim.x + threadIdx.x; i < n;
       i += gridDim.x * blockDim.x)
    out[i] = in[i];
}

// ---------------- one propagation round: gate-weighted segment max -----------
// One thread per (edge, 4-lane group): float4 source load, 4 atomics.
// All values are >= 0, so int atomicMax on float bits == float max.
__global__ void relax_kernel(const int* __restrict__ eidx, int E,
                             const float* __restrict__ ew,
                             const float* __restrict__ mIn,
                             float* __restrict__ mOut) {
  long long i = (long long)blockIdx.x * blockDim.x + threadIdx.x;
  long long total = (long long)E * 4;
  if (i >= total) return;
  int e = (int)(i >> 2);
  int q = (int)(i & 3) * 4;
  int s = eidx[e];
  int d = eidx[(long long)E + e];
  float w = ew[e];
  float4 mv = *(const float4*)&mIn[(long long)s * MDIM + q];
  int* o = (int*)&mOut[(long long)d * MDIM + q];
  atomicMax(o + 0, __float_as_int(mv.x * w));
  atomicMax(o + 1, __float_as_int(mv.y * w));
  atomicMax(o + 2, __float_as_int(mv.z * w));
  atomicMax(o + 3, __float_as_int(mv.w * w));
}

// ---------------- final elementwise max --------------------------------------
__global__ void final_max_kernel(const float* __restrict__ mask,
                                 const float* __restrict__ pos,
                                 const float* __restrict__ dom,
                                 float* __restrict__ out, int n) {
  for (int i = blockIdx.x * blockDim.x + threadIdx.x; i < n;
       i += gridDim.x * blockDim.x)
    out[i] = fmaxf(mask[i], fmaxf(pos[i], dom[i]));
}

extern "C" void kernel_launch(void* const* d_in, const int* in_sizes, int n_in,
                              void* d_out, int out_size, void* d_ws,
                              size_t ws_size, hipStream_t stream) {
  const float* x    = (const float*)d_in[0];
  const float* mask = (const float*)d_in[1];
  const int* se     = (const int*)d_in[2];
  const int* de     = (const int*)d_in[3];
  const float* sea  = (const float*)d_in[4];
  const float* dea  = (const float*)d_in[5];
  const float* Wtr  = (const float*)d_in[6];
  const float* btr  = (const float*)d_in[7];
  const float* Wpos = (const float*)d_in[8];
  const float* bpos = (const float*)d_in[9];
  const float* Wdom = (const float*)d_in[10];
  const float* bdom = (const float*)d_in[11];
  float* out = (float*)d_out;

  const int Nn = in_sizes[1] / MDIM;
  const int Es = in_sizes[2] / 2;
  const int Ed = in_sizes[3] / 2;
  const int nm = Nn * MDIM;

  char* ws = (char*)d_ws;
  size_t off = 0;
  auto walloc = [&](size_t bytes) -> void* {
    void* p = ws + off;
    off = (off + bytes + 255) & ~(size_t)255;
    return p;
  };
  _Float16* xh = (_Float16*)walloc((size_t)Nn * HDIM * sizeof(_Float16));
  _Float16* Wt = (_Float16*)walloc((size_t)2 * HDIM * TDIM * sizeof(_Float16));
  float* ewS  = (float*)walloc((size_t)Es * sizeof(float));
  float* ewD  = (float*)walloc((size_t)Ed * sizeof(float));
  float* bufA = (float*)walloc((size_t)nm * sizeof(float));
  float* bufB = (float*)walloc((size_t)nm * sizeof(float));
  float* bufC = (float*)walloc((size_t)nm * sizeof(float));
  float* bufD = (float*)walloc((size_t)nm * sizeof(float));

  // 1) node features -> f16 (L2-resident gather source)
  {
    int n = Nn * HDIM;
    cvt_f16_kernel<<<(n + 255) / 256, 256, 0, stream>>>(x, xh, n);
  }
  // 2) transpose W_trans to [T][2H] f16
  transpose_w_kernel<<<(2 * HDIM * TDIM + 255) / 256, 256, 0, stream>>>(Wtr, Wt);

  // 3) per-edge gate weights via WMMA (spatial: ad=4, dom: ad=1)
  {
    int tiles = (Es + 15) / 16;
    edge_gemm_kernel<<<(tiles + 7) / 8, 256, 0, stream>>>(xh, Wt, se, Es, sea,
                                                          4, Wpos, bpos, btr,
                                                          ewS);
  }
  {
    int tiles = (Ed + 15) / 16;
    edge_gemm_kernel<<<(tiles + 7) / 8, 256, 0, stream>>>(xh, Wt, de, Ed, dea,
                                                          1, Wdom, bdom, btr,
                                                          ewD);
  }

  // 4) spatial branch: 3 rounds of gate-weighted max propagation
  copy_f32_kernel<<<(nm + 255) / 256, 256, 0, stream>>>(mask, bufA, nm);
  {
    float* ping = bufA;
    float* pong = bufB;
    long long tot = (long long)Es * 4;
    int gb = (int)((tot + 255) / 256);
    for (int k = 0; k < 3; ++k) {
      copy_f32_kernel<<<(nm + 255) / 256, 256, 0, stream>>>(ping, pong, nm);
      relax_kernel<<<gb, 256, 0, stream>>>(se, Es, ewS, ping, pong);
      float* t = ping; ping = pong; pong = t;
    }
    // result in bufB (3 swaps: A->B, B->A, A->B)
  }

  // 5) dom branch
  copy_f32_kernel<<<(nm + 255) / 256, 256, 0, stream>>>(mask, bufC, nm);
  {
    float* ping = bufC;
    float* pong = bufD;
    long long tot = (long long)Ed * 4;
    int gb = (int)((tot + 255) / 256);
    for (int k = 0; k < 3; ++k) {
      copy_f32_kernel<<<(nm + 255) / 256, 256, 0, stream>>>(ping, pong, nm);
      relax_kernel<<<gb, 256, 0, stream>>>(de, Ed, ewD, ping, pong);
      float* t = ping; ping = pong; pong = t;
    }
    // result in bufD
  }

  // 6) final combine
  final_max_kernel<<<(nm + 255) / 256, 256, 0, stream>>>(mask, bufB, bufD, out,
                                                         nm);
}
